// TemporalGRU_12601434046568
// MI455X (gfx1250) — matrix operand order repaired and checked by reading, not compiled, not run on hardware
//
#include <hip/hip_runtime.h>

typedef _Float16 h16;
typedef __attribute__((ext_vector_type(16))) _Float16 v16h;
typedef __attribute__((ext_vector_type(8)))  _Float16 v8h;
typedef __attribute__((ext_vector_type(8)))  float    v8f;
typedef __attribute__((ext_vector_type(4)))  float    v4f;

#define B_DIM 1024
#define T_DIM 512
#define L2E   1.44269504088896340736f   // log2(e)

// One wave (32 lanes) owns a 16-row batch tile for one direction and runs the
// full T-step recurrence with weights resident in VGPRs.
//  - h(16x32) @ w_hh.T(32x96): 6x v_wmma_f32_16x16x32_f16 (K=32 == H, exact fit)
//  - x-projection: LAYER0 pads I=4 into K=32; LAYER1 uses 2 K-chunks of 32 (in=64)
// Gate tiles: n=0,1 -> r ; n=2,3 -> z ; n=4,5 -> candidate n.
// Accumulator chains start from WMMA inline SRC2=0. r/z biases fold into the
// exp2 argument FMA (pre-scaled by -log2e) -> zero extra VALU for bias.
template<int LAYER>
__global__ __launch_bounds__(32) void gru_kernel(
    const void* __restrict__ src_v,   // L0: float x[B][T][4] ; L1: h16 in[B][T][64]
    const float* __restrict__ w_ih,   // [2][96][IN_F]
    const float* __restrict__ w_hh,   // [2][96][32]
    const float* __restrict__ b_ih,   // [2][96]
    const float* __restrict__ b_hh,   // [2][96]
    h16* __restrict__ dst)            // [B][T][64], writes cols dir*32..dir*32+31
{
  constexpr int IN_F = (LAYER == 0) ? 4 : 64;
  constexpr int ICH  = (LAYER == 0) ? 1 : 2;

  const int dir  = blockIdx.x & 1;
  const int tile = blockIdx.x >> 1;          // 0..63
  const int b0   = tile * 16;
  const int lane = threadIdx.x;              // 0..31
  const int nn   = lane & 15;
  const bool hi  = lane >= 16;
  const int koff = hi ? 16 : 0;
  const int row  = hi ? lane - 16 : lane;    // A-operand row / M index

  __shared__ h16 lds_h[16][32];              // h staging (row-major) for A-layout reload

  if (!hi) {
    #pragma unroll
    for (int k = 0; k < 32; ++k) lds_h[nn][k] = (h16)0.f;
  }
  __syncthreads();

  // ---- preload weights (B-operand layout: lane N<16 = col N K0..15; lane N+16 = K16..31)
  v16h Bh[6];
  #pragma unroll
  for (int n = 0; n < 6; ++n) {
    const float* wr = w_hh + ((size_t)dir * 96 + (16 * n + nn)) * 32 + koff;
    v16h t;
    #pragma unroll
    for (int k = 0; k < 16; ++k) t[k] = (h16)wr[k];
    Bh[n] = t;
  }
  v16h Bi[ICH][6];
  #pragma unroll
  for (int c = 0; c < ICH; ++c)
    #pragma unroll
    for (int n = 0; n < 6; ++n) {
      const float* wr = w_ih + ((size_t)dir * 96 + (16 * n + nn)) * IN_F + c * 32 + koff;
      v16h t;
      #pragma unroll
      for (int k = 0; k < 16; ++k) {
        int f = c * 32 + koff + k;
        t[k] = (f < IN_F) ? (h16)wr[k] : (h16)0.f;   // zero-pad K beyond input dim
      }
      Bi[c][n] = t;
    }

  // per-lane biases (column g = 16n+nn)
  // r/z: pre-scale by -log2e so sigmoid arg becomes one FMA: fma(x, -L2E, nbrz)
  float nbrz[4], bxn[2], bhn[2];
  #pragma unroll
  for (int n = 0; n < 4; ++n)
    nbrz[n] = -L2E * (b_ih[dir * 96 + 16 * n + nn] + b_hh[dir * 96 + 16 * n + nn]);
  #pragma unroll
  for (int n = 0; n < 2; ++n) {
    bxn[n] = b_ih[dir * 96 + 64 + 16 * n + nn];
    bhn[n] = b_hh[dir * 96 + 64 + 16 * n + nn];
  }

  v8f hC[2];                                  // hidden state, f32 C-layout
  #pragma unroll
  for (int n = 0; n < 2; ++n)
    #pragma unroll
    for (int e = 0; e < 8; ++e) hC[n][e] = 0.f;

  const float* x32 = (const float*)src_v;
  const h16*   x16 = (const h16*)src_v;
  const h16 h0 = (h16)0.f;
  const v8f CZERO = {};                       // folds to WMMA inline SRC2 = 0

  #pragma unroll 1
  for (int s = 0; s < T_DIM; ++s) {
    const int tt = dir ? (T_DIM - 1 - s) : s;

    // ---- load x A-operands first (overlap global latency with hg WMMAs below)
    v16h Ax[ICH];
    #pragma unroll
    for (int c = 0; c < ICH; ++c) {
      if (LAYER == 0) {
        v4f xv = *(const v4f*)(x32 + (((size_t)(b0 + row)) * T_DIM + tt) * 4);
        v16h t;
        #pragma unroll
        for (int k = 0; k < 16; ++k) t[k] = h0;
        t[0] = hi ? h0 : (h16)xv[0];
        t[1] = hi ? h0 : (h16)xv[1];
        t[2] = hi ? h0 : (h16)xv[2];
        t[3] = hi ? h0 : (h16)xv[3];
        Ax[c] = t;
      } else {
        const h16* rp = x16 + (((size_t)(b0 + row)) * T_DIM + tt) * 64 + c * 32;
        v8h a0 = *(const v8h*)(rp + (hi ? 16 : 0));
        v8h a1 = *(const v8h*)(rp + (hi ? 24 : 8));
        Ax[c] = __builtin_shufflevector(a0, a1, 0, 1, 2, 3, 4, 5, 6, 7,
                                                 8, 9, 10, 11, 12, 13, 14, 15);
      }
    }

    // ---- A-operand for h from LDS (lane<16: K0..15; lane>=16: K16..31)
    v16h Ah;
    {
      const int c0 = hi ? 16 : 0, c1 = hi ? 24 : 8;
      v8h a0 = *(const v8h*)&lds_h[row][c0];
      v8h a1 = *(const v8h*)&lds_h[row][c1];
      Ah = __builtin_shufflevector(a0, a1, 0, 1, 2, 3, 4, 5, 6, 7,
                                           8, 9, 10, 11, 12, 13, 14, 15);
    }

    // ---- accumulator chains, each starting from inline-zero C
    v8f Crz[4], Cxn[2], Chn[2];
    #pragma unroll
    for (int n = 0; n < 4; ++n)
      Crz[n] = __builtin_amdgcn_wmma_f32_16x16x32_f16(false, Ah, false, Bh[n], (short)0, CZERO, false, false);
    #pragma unroll
    for (int n = 0; n < 2; ++n)
      Chn[n] = __builtin_amdgcn_wmma_f32_16x16x32_f16(false, Ah, false, Bh[4 + n], (short)0, CZERO, false, false);

    #pragma unroll
    for (int c = 0; c < ICH; ++c) {
      #pragma unroll
      for (int n = 0; n < 4; ++n)
        Crz[n] = __builtin_amdgcn_wmma_f32_16x16x32_f16(false, Ax[c], false, Bi[c][n], (short)0, Crz[n], false, false);
      #pragma unroll
      for (int n = 0; n < 2; ++n)
        Cxn[n] = (c == 0)
          ? __builtin_amdgcn_wmma_f32_16x16x32_f16(false, Ax[c], false, Bi[c][4 + n], (short)0, CZERO, false, false)
          : __builtin_amdgcn_wmma_f32_16x16x32_f16(false, Ax[c], false, Bi[c][4 + n], (short)0, Cxn[n], false, false);
    }

    __syncthreads();   // single-wave block: barrier lowers to S_NOP; orders LDS reuse

    // ---- gate epilogue: bias folded into exp2-arg FMAs; stage h_new to LDS
    #pragma unroll
    for (int n = 0; n < 2; ++n) {
      #pragma unroll
      for (int e = 0; e < 8; ++e) {
        float r = __builtin_amdgcn_rcpf(1.f + __builtin_amdgcn_exp2f(__builtin_fmaf(Crz[n][e],     -L2E, nbrz[n])));
        float z = __builtin_amdgcn_rcpf(1.f + __builtin_amdgcn_exp2f(__builtin_fmaf(Crz[2 + n][e], -L2E, nbrz[2 + n])));
        float u = __builtin_fmaf(r, Chn[n][e] + bhn[n], Cxn[n][e] + bxn[n]);
        float g = __builtin_fmaf(2.f, __builtin_amdgcn_rcpf(1.f + __builtin_amdgcn_exp2f(u * (-2.f * L2E))), -1.f);
        float h = __builtin_fmaf(z, hC[n][e] - g, g);      // (1-z)*g + z*h_old
        hC[n][e] = h;
        lds_h[e + (hi ? 8 : 0)][16 * n + nn] = (h16)h;
      }
    }
    __syncthreads();

    // ---- coalesced h_out store: each lane streams 16B x2 of one row from LDS
    {
      const int sr = lane >> 1;               // 0..15  (batch row within tile)
      const int sc = (lane & 1) * 16;         // 0 or 16 (hidden col base)
      v8h s0 = *(const v8h*)&lds_h[sr][sc];
      v8h s1 = *(const v8h*)&lds_h[sr][sc + 8];
      h16* dp = dst + (((size_t)(b0 + sr)) * T_DIM + tt) * 64 + dir * 32 + sc;
      *(v8h*)dp = s0;
      *(v8h*)(dp + 8) = s1;
    }
  }
}

// Attention pooling + FC + sigmoid. One block per batch row.
__global__ __launch_bounds__(256) void attn_fc_kernel(
    const h16* __restrict__ out1,           // [B][T][64] f16
    const float* __restrict__ attn_w,       // [64]
    const float* __restrict__ attn_b,       // [1]
    const float* __restrict__ fc_w,         // [64]
    const float* __restrict__ fc_b,         // [1]
    float* __restrict__ out)                // [B]
{
  const int b   = blockIdx.x;
  const int tid = threadIdx.x;
  __shared__ float logit[T_DIM];
  __shared__ float red[256];
  __shared__ float ctx[64];
  const h16* base = out1 + (size_t)b * T_DIM * 64;

  for (int t = tid; t < T_DIM; t += 256) {
    float acc = attn_b[0];
    #pragma unroll 8
    for (int f = 0; f < 64; ++f) acc += (float)base[(size_t)t * 64 + f] * attn_w[f];
    logit[t] = acc;
  }
  __syncthreads();

  float m = -3.402823466e+38f;
  for (int t = tid; t < T_DIM; t += 256) m = fmaxf(m, logit[t]);
  red[tid] = m; __syncthreads();
  for (int s = 128; s > 0; s >>= 1) { if (tid < s) red[tid] = fmaxf(red[tid], red[tid + s]); __syncthreads(); }
  m = red[0]; __syncthreads();

  float e = 0.f;
  for (int t = tid; t < T_DIM; t += 256) {
    float v = __builtin_amdgcn_exp2f((logit[t] - m) * L2E);
    logit[t] = v; e += v;
  }
  red[tid] = e; __syncthreads();
  for (int s = 128; s > 0; s >>= 1) { if (tid < s) red[tid] += red[tid + s]; __syncthreads(); }
  const float inv = __builtin_amdgcn_rcpf(red[0]);
  __syncthreads();

  if (tid < 64) {
    float acc = 0.f;
    for (int t = 0; t < T_DIM; ++t) acc += logit[t] * (float)base[(size_t)t * 64 + tid];
    ctx[tid] = acc * inv;
  }
  __syncthreads();

  if (tid == 0) {
    float acc = fc_b[0];
    #pragma unroll 8
    for (int f = 0; f < 64; ++f) acc += ctx[f] * fc_w[f];
    out[b] = __builtin_amdgcn_rcpf(1.f + __builtin_amdgcn_exp2f(-acc * L2E));
  }
}

extern "C" void kernel_launch(void* const* d_in, const int* in_sizes, int n_in,
                              void* d_out, int out_size, void* d_ws, size_t ws_size,
                              hipStream_t stream) {
  (void)in_sizes; (void)n_in; (void)out_size; (void)ws_size;
  const float* x      = (const float*)d_in[0];
  const float* w_ih0  = (const float*)d_in[1];
  const float* w_hh0  = (const float*)d_in[2];
  const float* b_ih0  = (const float*)d_in[3];
  const float* b_hh0  = (const float*)d_in[4];
  const float* w_ih1  = (const float*)d_in[5];
  const float* w_hh1  = (const float*)d_in[6];
  const float* b_ih1  = (const float*)d_in[7];
  const float* b_hh1  = (const float*)d_in[8];
  const float* attn_w = (const float*)d_in[9];
  const float* attn_b = (const float*)d_in[10];
  const float* fc_w   = (const float*)d_in[11];
  const float* fc_b   = (const float*)d_in[12];
  float* out = (float*)d_out;

  h16* ws0 = (h16*)d_ws;                                  // layer-0 output, f16 [B][T][64]
  h16* ws1 = ws0 + (size_t)B_DIM * T_DIM * 64;            // layer-1 output, f16 [B][T][64]

  dim3 grid(2 * (B_DIM / 16));                            // dir x batch-tile, 1 wave each
  gru_kernel<0><<<grid, 32, 0, stream>>>((const void*)x,   w_ih0, w_hh0, b_ih0, b_hh0, ws0);
  gru_kernel<1><<<grid, 32, 0, stream>>>((const void*)ws0, w_ih1, w_hh1, b_ih1, b_hh1, ws1);
  attn_fc_kernel<<<B_DIM, 256, 0, stream>>>(ws1, attn_w, attn_b, fc_w, fc_b, out);
}
